// LatentRelMM_88983132439284
// MI455X (gfx1250) — matrix operand order repaired and checked by reading, not compiled, not run on hardware
//
#include <hip/hip_runtime.h>
#include <hip/hip_bf16.h>

// ---------------------------------------------------------------------------
// Fused LatentRelMM kernel for MI455X (gfx1250, wave32, WMMA bf16)
// B=65536, H=1024, N=256 (memory slots)
// One wave owns 32 batch rows (2 A-fragments) -> 2 WMMAs per B-fragment load.
// B fragments are manually double-buffered 2 deep to avoid full loadcnt drains.
// ---------------------------------------------------------------------------

typedef __attribute__((ext_vector_type(16))) __bf16 v16bf;
typedef __attribute__((ext_vector_type(8)))  float  v8f;

#define HDIM 1024
#define NSLOT 256
#define ROWS_PER_WAVE 32
#define WAVES_PER_BLOCK 2
#define ROWS_PER_BLOCK (ROWS_PER_WAVE * WAVES_PER_BLOCK)   // 64
#define LDS_ROW_STRIDE 264   // 256 bf16 + 8 pad -> 528B row stride, conflict free

// Native f32 -> bf16 (fptrunc, RNE): clang emits v_cvt_pk_bf16_f32 pairs.
__device__ __forceinline__ __bf16 f2bf(float f) { return (__bf16)f; }

#define WMMA_BF16(A, B, C) __builtin_amdgcn_wmma_f32_16x16x32_bf16( \
    false, (A), false, (B), (short)0, (C), false, false)

// ---------------------------------------------------------------------------
// Prep: pack memory (f32 [N, H]) into per-lane WMMA B-fragment bf16 layout.
// B fragment (32K x 16N, bf16): lanes 0-15 -> n=lane,   K = kbase+0..7,16..23
//                               lanes 16-31-> n=lane-16,K = kbase+8..15,24..31
// pb2: B = memory^T  (scores   s @ mem^T): frag(kc,nt): B[k,n]=mem[n0+n][k0+k]
// pb3: B = memory    (rel   attn @ mem  ): frag(ht,kc): B[k,n]=mem[k0+k][h0+n]
// Each fragment = 32 lanes * 16 bf16 = 1024 B, per-lane 32B contiguous.
// ---------------------------------------------------------------------------
__global__ void latentrel_prep(const float* __restrict__ mem,
                               __bf16* __restrict__ pb2,
                               __bf16* __restrict__ pb3) {
    int tid  = blockIdx.x * blockDim.x + threadIdx.x;
    int lane = tid & 31;
    int frag = tid >> 5;
    int ln   = lane & 15;
    int koff = (lane >> 4) * 8;
    if (frag < 512) {                       // pb2: 32 k-chunks x 16 n-tiles
        int kc = frag >> 4, nt = frag & 15;
        int n  = nt * 16 + ln;
        int kb = kc * 32 + koff;
        __bf16* dst = pb2 + (size_t)frag * 512 + lane * 16;
        #pragma unroll
        for (int j = 0; j < 16; ++j)
            dst[j] = f2bf(mem[(size_t)n * HDIM + kb + (j & 7) + ((j >> 3) << 4)]);
    } else if (frag < 1024) {               // pb3: 64 h-tiles x 8 k-chunks
        int f  = frag - 512;
        int ht = f >> 3, kc = f & 7;
        int col = ht * 16 + ln;
        int kb  = kc * 32 + koff;
        __bf16* dst = pb3 + (size_t)f * 512 + lane * 16;
        #pragma unroll
        for (int j = 0; j < 16; ++j)
            dst[j] = f2bf(mem[(size_t)(kb + (j & 7) + ((j >> 3) << 4)) * HDIM + col]);
    }
}

// ---------------------------------------------------------------------------
// Final deterministic loss reduction: fixed-schedule sum of block partials.
// ---------------------------------------------------------------------------
__global__ void latentrel_loss_reduce(const float* __restrict__ partials,
                                      int n, float invB,
                                      float* __restrict__ out) {
    __shared__ float red[256];
    float s = 0.0f;
    for (int i = threadIdx.x; i < n; i += 256) s += partials[i];
    red[threadIdx.x] = s;
    __syncthreads();
    #pragma unroll
    for (int stride = 128; stride > 0; stride >>= 1) {
        if (threadIdx.x < stride) red[threadIdx.x] += red[threadIdx.x + stride];
        __syncthreads();
    }
    if (threadIdx.x == 0) out[0] = red[0] * invB;
}

// Build one A fragment (16 rows x 32 K, bf16 s = u*m) for this k-chunk.
__device__ __forceinline__ v16bf build_sfrag(const float* __restrict__ up,
                                             const float* __restrict__ mp,
                                             int kb) {
    float us[16], ms[16];
    *(float4*)&us[0]  = *(const float4*)(up + kb);
    *(float4*)&us[4]  = *(const float4*)(up + kb + 4);
    *(float4*)&us[8]  = *(const float4*)(up + kb + 16);
    *(float4*)&us[12] = *(const float4*)(up + kb + 20);
    *(float4*)&ms[0]  = *(const float4*)(mp + kb);
    *(float4*)&ms[4]  = *(const float4*)(mp + kb + 4);
    *(float4*)&ms[8]  = *(const float4*)(mp + kb + 16);
    *(float4*)&ms[12] = *(const float4*)(mp + kb + 20);
    v16bf a;
    #pragma unroll
    for (int j = 0; j < 16; ++j) a[j] = f2bf(us[j] * ms[j]);
    return a;
}

// Row-wise softmax over one 16-row accumulator set; stores unnormalized
// exp() to LDS (bf16) and returns 1/rowsum per (r, half) lane mapping.
__device__ __forceinline__ void softmax_store(const v8f* acc,
                                              __bf16* ldsBase,  // row 0 of set
                                              int half, int ln,
                                              float* rinv) {
    float mx[8];
    #pragma unroll
    for (int r = 0; r < 8; ++r) {
        float v = acc[0][r];
        #pragma unroll
        for (int t = 1; t < 16; ++t) v = fmaxf(v, acc[t][r]);
        #pragma unroll
        for (int d = 1; d < 16; d <<= 1) v = fmaxf(v, __shfl_xor(v, d, 32));
        mx[r] = v;
    }
    float rs[8];
    #pragma unroll
    for (int r = 0; r < 8; ++r) rs[r] = 0.0f;
    #pragma unroll
    for (int t = 0; t < 16; ++t) {
        #pragma unroll
        for (int r = 0; r < 8; ++r) {
            float p = __expf(acc[t][r] - mx[r]);       // unnormalized attn
            rs[r] += p;
            ldsBase[(r + half * 8) * LDS_ROW_STRIDE + t * 16 + ln] = f2bf(p);
        }
    }
    #pragma unroll
    for (int r = 0; r < 8; ++r) {
        float v = rs[r];
        #pragma unroll
        for (int d = 1; d < 16; d <<= 1) v += __shfl_xor(v, d, 32);
        rinv[r] = 1.0f / v;                            // denom folded into rel
    }
}

// ---------------------------------------------------------------------------
// Main fused kernel: one wave handles 32 batch rows end-to-end.
// ---------------------------------------------------------------------------
__global__ __launch_bounds__(WAVES_PER_BLOCK * 32)
void latentrel_fused(const float* __restrict__ user,
                     const float* __restrict__ music,
                     const float* __restrict__ W_out,
                     const float* __restrict__ b_out,
                     const int*   __restrict__ label,
                     const __bf16* __restrict__ pb2,
                     const __bf16* __restrict__ pb3,
                     float* __restrict__ out,
                     float* __restrict__ partials) {
    __shared__ __bf16 attnLds[WAVES_PER_BLOCK][32 * LDS_ROW_STRIDE];
    __shared__ float  lossRed[WAVES_PER_BLOCK * 2];

    const int lane    = threadIdx.x & 31;
    const int wave    = threadIdx.x >> 5;
    const int ln      = lane & 15;
    const int half    = lane >> 4;
    const int rowBase = blockIdx.x * ROWS_PER_BLOCK + wave * ROWS_PER_WAVE;

    // A-fragment lane mapping (ISA 16-bit A 16x32): lane holds one row,
    // elements 0..7 -> K = kb..kb+7, elements 8..15 -> K = kb+16..kb+23,
    // kb = kc*32 + half*8. Row set 0: rows rowBase+ln; set 1: rowBase+16+ln.
    const float* up0 = user  + (size_t)(rowBase + ln) * HDIM;
    const float* mp0 = music + (size_t)(rowBase + ln) * HDIM;
    const float* up1 = up0 + (size_t)16 * HDIM;
    const float* mp1 = mp0 + (size_t)16 * HDIM;

    const v16bf* pb2v = (const v16bf*)pb2;
    const v16bf* pb3v = (const v16bf*)pb3;

    // ---------------- Phase 2: scores = (u*m) @ memory^T  [32 x 256] -------
    v8f acc0[16], acc1[16];
    #pragma unroll
    for (int t = 0; t < 16; ++t) { acc0[t] = 0.0f; acc1[t] = 0.0f; }

    // 2-deep rotating B-fragment buffers over the flat 512-fragment stream.
    v16bf bA = pb2v[lane];
    v16bf bB = pb2v[32 + lane];
    for (int kc = 0; kc < HDIM / 32; ++kc) {
        int kb = kc * 32 + half * 8;
        v16bf a0 = build_sfrag(up0, mp0, kb);
        v16bf a1 = build_sfrag(up1, mp1, kb);
        #pragma unroll
        for (int t = 0; t < 16; t += 2) {
            int f = kc * 16 + t;
            v16bf bn0 = pb2v[(size_t)((f + 2) & 511) * 32 + lane];
            v16bf bn1 = pb2v[(size_t)((f + 3) & 511) * 32 + lane];
            acc0[t]     = WMMA_BF16(a0, bA, acc0[t]);
            acc1[t]     = WMMA_BF16(a1, bA, acc1[t]);
            acc0[t + 1] = WMMA_BF16(a0, bB, acc0[t + 1]);
            acc1[t + 1] = WMMA_BF16(a1, bB, acc1[t + 1]);
            bA = bn0;
            bB = bn1;
        }
    }

    // ---------------- Softmax (row-wise over N=256), both row sets ----------
    float rinv0[8], rinv1[8];
    softmax_store(acc0, &attnLds[wave][0],                   half, ln, rinv0);
    softmax_store(acc1, &attnLds[wave][16 * LDS_ROW_STRIDE], half, ln, rinv1);
    __syncthreads();

    // Reload attn as A fragments (LDS transpose), K = 256 in 8 chunks of 32.
    v16bf af0[8], af1[8];
    #pragma unroll
    for (int kc = 0; kc < 8; ++kc) {
        int kb = kc * 32 + half * 8;
        union { uint4 q[2]; v16bf v; } t0, t1;
        t0.q[0] = *(const uint4*)&attnLds[wave][ln * LDS_ROW_STRIDE + kb];
        t0.q[1] = *(const uint4*)&attnLds[wave][ln * LDS_ROW_STRIDE + kb + 16];
        t1.q[0] = *(const uint4*)&attnLds[wave][(16 + ln) * LDS_ROW_STRIDE + kb];
        t1.q[1] = *(const uint4*)&attnLds[wave][(16 + ln) * LDS_ROW_STRIDE + kb + 16];
        af0[kc] = t0.v;
        af1[kc] = t1.v;
    }

    // ---------------- Phase 3: rel = attn @ memory, fused diff/head ---------
    float ssq[16], o0a[16], o1a[16];
    #pragma unroll
    for (int r = 0; r < 16; ++r) { ssq[r] = 0.0f; o0a[r] = 0.0f; o1a[r] = 0.0f; }

    bA = pb3v[lane];
    bB = pb3v[32 + lane];
    for (int ht = 0; ht < HDIM / 16; ++ht) {
        v8f d0 = 0.0f, d1 = 0.0f;
        #pragma unroll
        for (int kc = 0; kc < 8; kc += 2) {
            int f = ht * 8 + kc;
            v16bf bn0 = pb3v[(size_t)((f + 2) & 511) * 32 + lane];
            v16bf bn1 = pb3v[(size_t)((f + 3) & 511) * 32 + lane];
            d0 = WMMA_BF16(af0[kc], bA, d0);
            d1 = WMMA_BF16(af1[kc], bA, d1);
            d0 = WMMA_BF16(af0[kc + 1], bB, d0);
            d1 = WMMA_BF16(af1[kc + 1], bB, d1);
            bA = bn0;
            bB = bn1;
        }
        int h = ht * 16 + ln;
        float2 w = *(const float2*)(W_out + h * 2);
        #pragma unroll
        for (int r = 0; r < 8; ++r) {
            // row set 0
            size_t i0 = (size_t)(rowBase + r + half * 8) * HDIM + h;
            float u  = user[i0];
            float mu = music[i0];
            float dv = u + d0[r] * rinv0[r] - mu;
            ssq[r] = fmaf(dv, dv, ssq[r]);
            float s = u * mu;
            o0a[r] = fmaf(s, w.x, o0a[r]);
            o1a[r] = fmaf(s, w.y, o1a[r]);
            // row set 1
            size_t i1 = i0 + (size_t)16 * HDIM;
            float u2  = user[i1];
            float mu2 = music[i1];
            float dv2 = u2 + d1[r] * rinv1[r] - mu2;
            ssq[8 + r] = fmaf(dv2, dv2, ssq[8 + r]);
            float s2 = u2 * mu2;
            o0a[8 + r] = fmaf(s2, w.x, o0a[8 + r]);
            o1a[8 + r] = fmaf(s2, w.y, o1a[8 + r]);
        }
    }

    // ---------------- Reductions + outputs ----------------------------------
    #pragma unroll
    for (int r = 0; r < 16; ++r) {
        float a = ssq[r], b = o0a[r], c = o1a[r];
        #pragma unroll
        for (int d = 1; d < 16; d <<= 1) {
            a += __shfl_xor(a, d, 32);
            b += __shfl_xor(b, d, 32);
            c += __shfl_xor(c, d, 32);
        }
        ssq[r] = a; o0a[r] = b; o1a[r] = c;
    }
    if (ln == 0) {                 // lanes 0 and 16 each finalize 16 rows
        float b0 = b_out[0], b1 = b_out[1];
        float lsum = 0.0f;
        #pragma unroll
        for (int rr = 0; rr < 16; ++rr) {
            int r   = rr & 7;
            int set = rr >> 3;
            int row = rowBase + set * 16 + r + half * 8;
            float score = sqrtf(ssq[rr]);
            float msk = (float)(2 * label[row] - 1);
            lsum += msk * score;
            out[1 + row * 2]     = o0a[rr] + b0;
            out[1 + row * 2 + 1] = o1a[rr] + b1;
        }
        lossRed[wave * 2 + half] = lsum;           // fixed-order block partial
    }
    __syncthreads();
    if (threadIdx.x == 0) {
        float s = 0.0f;
        #pragma unroll
        for (int i = 0; i < WAVES_PER_BLOCK * 2; ++i) s += lossRed[i];
        partials[blockIdx.x] = s;
    }
}

// ---------------------------------------------------------------------------
extern "C" void kernel_launch(void* const* d_in, const int* in_sizes, int n_in,
                              void* d_out, int out_size, void* d_ws, size_t ws_size,
                              hipStream_t stream) {
    const float* user   = (const float*)d_in[0];
    const float* music  = (const float*)d_in[1];
    const float* memory = (const float*)d_in[2];
    const float* W_out  = (const float*)d_in[3];
    const float* b_out  = (const float*)d_in[4];
    const int*   label  = (const int*)d_in[5];
    float* out = (float*)d_out;

    const int B = in_sizes[0] / HDIM;
    const int nBlocks = B / ROWS_PER_BLOCK;

    __bf16* pb2 = (__bf16*)d_ws;                          // 512 KB
    __bf16* pb3 = pb2 + (size_t)512 * 512;                // 512 KB
    float* partials = (float*)(pb3 + (size_t)512 * 512);  // nBlocks * 4 B

    latentrel_prep<<<128, 256, 0, stream>>>(memory, pb2, pb3);
    latentrel_fused<<<nBlocks, WAVES_PER_BLOCK * 32, 0, stream>>>(
        user, music, W_out, b_out, label, pb2, pb3, out, partials);
    latentrel_loss_reduce<<<1, 256, 0, stream>>>(partials, nBlocks,
                                                 1.0f / (float)B, out);
}
